// DSFDecoder_53137335386844
// MI455X (gfx1250) — matrix operand order repaired and checked by reading, not compiled, not run on hardware
//
#include <hip/hip_runtime.h>
#include <hip/hip_bf16.h>
#include <math.h>

// ---------------------------------------------------------------------------
// CDNA5 WMMA (wave32): v_wmma_f32_16x16x32_bf16, D = A(16x32) x B(32x16) + C
// ---------------------------------------------------------------------------
typedef __attribute__((ext_vector_type(16))) __bf16 v16bf;
typedef __attribute__((ext_vector_type(8)))  float  v8f;

union FragAB {
    v16bf          v;
    unsigned short u[16];
    uint4          q[2];
};

__device__ __forceinline__ unsigned short f2bf(float f) {
    union { float f; unsigned u; } cv; cv.f = f;
    unsigned u = cv.u;
    if ((u & 0x7fffffffu) > 0x7f800000u)            // NaN -> quiet NaN bf16
        return (unsigned short)((u >> 16) | 0x40u);
    return (unsigned short)((u + 0x7fffu + ((u >> 16) & 1u)) >> 16);  // RNE
}

// ---------------------------------------------------------------------------
// NHWC implicit-GEMM conv (1x1 or 3x3 pad=1), bf16 in / f32 accum, LDS-free.
//   Act : [nb][P][Cin]        activations bf16 (NHWC), P = H*W
//   Wt  : [Cout][TAPS][Cin]   weights bf16
// One wave per block. Wave tile: 16 Cout x (16*NSUB) pixels.
//   WMMA A = weights  (M = cout,  two contiguous 16B loads per lane)
//   WMMA B = act      (N = pixel, two 16B loads per lane, tap = shift)
// Outputs (any subset): f32 NCHW (final), f32 NHWC, bf16 NHWC.
// ---------------------------------------------------------------------------
template <int TAPS, int NSUB>
__global__ __launch_bounds__(32) void conv_wmma_nhwc(
    const unsigned short* __restrict__ Act,
    const unsigned short* __restrict__ Wt,
    const float* __restrict__ bias,
    const float* __restrict__ residNHWC,     // nullable, [nb][P][Cout] f32
    float* __restrict__ outF_nchw,           // nullable, [nb][Cout][P] f32
    float* __restrict__ outF_nhwc,           // nullable, [nb][P][Cout] f32
    unsigned short* __restrict__ outB_nhwc,  // nullable, [nb][P][Cout] bf16
    int Cin, int Cout, int P, int logW, int H, int relu)
{
    const int lane = threadIdx.x;
    const int half = lane & 15;
    const int hi   = lane >> 4;
    const int m0   = blockIdx.y * 16;              // cout tile base
    const int pb   = blockIdx.x * (16 * NSUB);     // pixel tile base (same row)
    const int b    = blockIdx.z;
    const int W    = 1 << logW;
    const int h0   = pb >> logW;
    const int wb   = pb & (W - 1);

    const unsigned short* AB   = Act + (size_t)b * (size_t)P * (size_t)Cin;
    const unsigned short* Wrow = Wt + (size_t)(m0 + half) * TAPS * (size_t)Cin;

    v8f acc[NSUB];
    #pragma unroll
    for (int s = 0; s < NSUB; ++s) acc[s] = (v8f){};

    #pragma unroll
    for (int t = 0; t < TAPS; ++t) {
        const int kh = (TAPS == 1) ? 1 : (t / 3);
        const int kw = (TAPS == 1) ? 1 : (t % 3);
        const int d  = (kh - 1) * W + (kw - 1);

        bool ok[NSUB];
        const unsigned short* bp[NSUB];
        #pragma unroll
        for (int s = 0; s < NSUB; ++s) {
            if (TAPS == 1) {
                ok[s] = true;
            } else {
                const bool rowOK = (unsigned)(h0 + kh - 1) < (unsigned)H;
                const int  wp    = wb + s * 16 + half + (kw - 1);
                ok[s] = rowOK && ((unsigned)wp < (unsigned)W);
            }
            bp[s] = AB + (size_t)(pb + s * 16 + half + d) * (size_t)Cin;
        }
        const unsigned short* aptr = Wrow + t * Cin;

        for (int k0 = 0; k0 < Cin; k0 += 32) {
            // A fragment (weights): K = (i>>3)*16 + hi*8 + (i&7)
            FragAB fa;
            fa.q[0] = *(const uint4*)(aptr + k0 + hi * 8);
            fa.q[1] = *(const uint4*)(aptr + k0 + 16 + hi * 8);
            if (k0 + 32 < Cin)
                __builtin_prefetch(aptr + k0 + 32, 0, 1);

            #pragma unroll
            for (int s = 0; s < NSUB; ++s) {
                // B fragment (activations): K = (i>>3)*16 + hi*8 + (i&7)
                FragAB fb;
                if (ok[s]) {
                    const unsigned short* src = bp[s] + k0 + hi * 8;
                    fb.q[0] = *(const uint4*)(src);
                    fb.q[1] = *(const uint4*)(src + 16);
                } else {
                    uint4 z = {0u, 0u, 0u, 0u};
                    fb.q[0] = z; fb.q[1] = z;
                }
                acc[s] = __builtin_amdgcn_wmma_f32_16x16x32_bf16(
                             false, fa.v, false, fb.v, (short)0, acc[s],
                             false, false);
            }
        }
    }

    // ---- epilogue: VGPR r -> cout m0 + hi*8 + r, pixel pb + s*16 + half ----
    const int mb = m0 + hi * 8;
    const float4 bq0 = *(const float4*)(bias + mb);
    const float4 bq1 = *(const float4*)(bias + mb + 4);
    const float bv[8] = {bq0.x, bq0.y, bq0.z, bq0.w, bq1.x, bq1.y, bq1.z, bq1.w};

    #pragma unroll
    for (int s = 0; s < NSUB; ++s) {
        const size_t pix  = (size_t)(pb + s * 16 + half);
        const size_t base = ((size_t)b * P + pix) * (size_t)Cout + mb;

        float v[8];
        #pragma unroll
        for (int r = 0; r < 8; ++r) v[r] = acc[s][r] + bv[r];

        if (residNHWC) {
            const float4 r0 = ((const float4*)(residNHWC + base))[0];
            const float4 r1 = ((const float4*)(residNHWC + base))[1];
            v[0] += r0.x; v[1] += r0.y; v[2] += r0.z; v[3] += r0.w;
            v[4] += r1.x; v[5] += r1.y; v[6] += r1.z; v[7] += r1.w;
        }
        if (relu) {
            #pragma unroll
            for (int r = 0; r < 8; ++r) v[r] = fmaxf(v[r], 0.f);
        }
        if (outF_nhwc) {
            ((float4*)(outF_nhwc + base))[0] = make_float4(v[0], v[1], v[2], v[3]);
            ((float4*)(outF_nhwc + base))[1] = make_float4(v[4], v[5], v[6], v[7]);
        }
        if (outB_nhwc) {
            union { unsigned short us[8]; uint4 q; } pk;
            #pragma unroll
            for (int r = 0; r < 8; ++r) pk.us[r] = f2bf(v[r]);
            *(uint4*)(outB_nhwc + base) = pk.q;
        }
        if (outF_nchw) {   // final output: per-r stores are lane-coalesced
            #pragma unroll
            for (int r = 0; r < 8; ++r)
                outF_nchw[((size_t)b * Cout + mb + r) * (size_t)P + pix] = v[r];
        }
    }
}

// ---------------------------------------------------------------------------
// Dynamic-sparse-conv condition net + apply (B=8, C=256, H=W=64, NCHW f32 in)
// ---------------------------------------------------------------------------
__global__ void mean_hw_kernel(const float* __restrict__ x, float* __restrict__ gm)
{
    __shared__ float red[256];
    const int bc = blockIdx.x;                      // 0..2047
    const float* p = x + ((size_t)bc << 12);
    float s = 0.f;
    for (int i = threadIdx.x; i < 4096; i += 256) s += p[i];
    red[threadIdx.x] = s;
    __syncthreads();
    for (int o = 128; o > 0; o >>= 1) {
        if (threadIdx.x < o) red[threadIdx.x] += red[threadIdx.x + o];
        __syncthreads();
    }
    if (threadIdx.x == 0) gm[bc] = red[0] * (1.f / 4096.f);
}

__global__ void cond_fc1_gelu(const float* __restrict__ gm,
                              const float* __restrict__ w1,
                              const float* __restrict__ b1,
                              float* __restrict__ hbuf)
{
    const int id = blockIdx.x * blockDim.x + threadIdx.x;
    if (id >= 8 * 64) return;
    const int b = id >> 6, j = id & 63;
    const float* g = gm + b * 256;
    const float* w = w1 + j * 256;
    float s = b1[j];
    for (int k = 0; k < 256; ++k) s += g[k] * w[k];
    hbuf[id] = 0.5f * s * (1.f + erff(s * 0.70710678118654752f));  // exact GELU
}

__global__ void cond_fc2_softmax(const float* __restrict__ hbuf,
                                 const float* __restrict__ w2,
                                 const float* __restrict__ b2,
                                 float* __restrict__ wts)
{
    const int id = blockIdx.x * blockDim.x + threadIdx.x;
    if (id >= 8 * 256) return;
    const int b = id >> 8, c = id & 255;
    const float* h = hbuf + b * 64;
    float v[9];
    for (int t = 0; t < 9; ++t) {
        const float* w = w2 + (size_t)(c * 9 + t) * 64;
        float s = b2[c * 9 + t];
        for (int k = 0; k < 64; ++k) s += h[k] * w[k];
        v[t] = s;
    }
    float mx = v[0];
    for (int t = 1; t < 9; ++t) mx = fmaxf(mx, v[t]);
    float se = 0.f;
    for (int t = 0; t < 9; ++t) { v[t] = expf(v[t] - mx); se += v[t]; }
    const float inv = 1.f / se;
    float* o = wts + (size_t)(b * 256 + c) * 9;
    for (int t = 0; t < 9; ++t) o[t] = v[t] * inv;
}

// y = depthwise3x3(x; wts[b,c,:]) + x ; optional relu
// outF: NCHW f32 [8][256][4096] ; outB: NHWC bf16 [8][4096][CoutTot] at chOff
__global__ void dsc_apply(const float* __restrict__ x,
                          const float* __restrict__ wts,
                          float* __restrict__ outF,
                          unsigned short* __restrict__ outB,
                          int chOff, int CoutTot, int relu)
{
    const size_t total = (size_t)8 * 256 * 4096;
    for (size_t idx = (size_t)blockIdx.x * blockDim.x + threadIdx.x;
         idx < total; idx += (size_t)gridDim.x * blockDim.x) {
        const int p = (int)(idx & 4095);
        const int c = (int)((idx >> 12) & 255);
        const int b = (int)(idx >> 20);
        const int h = p >> 6, w = p & 63;
        const float* wp = wts + (size_t)(b * 256 + c) * 9;
        const float* xp = x + ((size_t)(b * 256 + c) << 12);
        float s = 0.f;
        #pragma unroll
        for (int kh = 0; kh < 3; ++kh) {
            const int hh = h + kh - 1;
            if (hh < 0 || hh >= 64) continue;
            #pragma unroll
            for (int kw = 0; kw < 3; ++kw) {
                const int ww = w + kw - 1;
                if (ww < 0 || ww >= 64) continue;
                s += wp[kh * 3 + kw] * xp[hh * 64 + ww];
            }
        }
        float y = s + xp[p];
        if (relu) y = fmaxf(y, 0.f);
        if (outF) outF[idx] = y;
        if (outB) outB[((size_t)b * 4096 + p) * CoutTot + chOff + c] = f2bf(y);
    }
}

// pixel_shuffle(2), NHWC: in [8][4096][1024] -> out [8][16384][256] (bf16)
__global__ void pixshuf_nhwc(const unsigned short* __restrict__ in,
                             unsigned short* __restrict__ out)
{
    const size_t total = (size_t)8 * 16384 * 256;
    for (size_t idx = (size_t)blockIdx.x * blockDim.x + threadIdx.x;
         idx < total; idx += (size_t)gridDim.x * blockDim.x) {
        const int c  = (int)(idx & 255);
        const int p  = (int)((idx >> 8) & 16383);
        const int b  = (int)(idx >> 22);
        const int oh = p >> 7, ow = p & 127;
        const int c4 = c * 4 + (oh & 1) * 2 + (ow & 1);
        const int ip = (oh >> 1) * 64 + (ow >> 1);
        out[idx] = in[((size_t)b * 4096 + ip) * 1024 + c4];
    }
}

// weights: f32 [Cout][Cin][kh][kw] -> bf16 [Cout][taps][Cin]
__global__ void wprep_kernel(const float* __restrict__ src,
                             unsigned short* __restrict__ dst,
                             int Cout, int Cin, int taps)
{
    const size_t total = (size_t)Cout * taps * Cin;
    const size_t idx = (size_t)blockIdx.x * blockDim.x + threadIdx.x;
    if (idx >= total) return;
    const int ci   = (int)(idx % Cin);
    const int rest = (int)(idx / Cin);
    const int t    = rest % taps;
    const int co   = rest / taps;
    dst[idx] = f2bf(src[((size_t)co * Cin + ci) * taps + t]);
}

// ---------------------------------------------------------------------------
extern "C" void kernel_launch(void* const* d_in, const int* in_sizes, int n_in,
                              void* d_out, int out_size, void* d_ws, size_t ws_size,
                              hipStream_t stream)
{
    const float* x1      = (const float*)d_in[0];
    const float* x2      = (const float*)d_in[1];
    const float* dsc_w1[2] = { (const float*)d_in[2], (const float*)d_in[6] };
    const float* dsc_b1[2] = { (const float*)d_in[3], (const float*)d_in[7] };
    const float* dsc_w2[2] = { (const float*)d_in[4], (const float*)d_in[8] };
    const float* dsc_b2[2] = { (const float*)d_in[5], (const float*)d_in[9] };
    const float* align_w = (const float*)d_in[10];
    const float* align_b = (const float*)d_in[11];
    const float* up_w1   = (const float*)d_in[12];
    const float* up_b1   = (const float*)d_in[13];
    const float* up_w2   = (const float*)d_in[14];
    const float* up_b2   = (const float*)d_in[15];
    const float* re_w1   = (const float*)d_in[16];
    const float* re_b1   = (const float*)d_in[17];
    const float* re_w2   = (const float*)d_in[18];
    const float* re_b2   = (const float*)d_in[19];

    // ---- workspace carve-up (aliased by lifetime; ~190 MB total) ----
    char* ws = (char*)d_ws;
    size_t off = 0;
    auto alloc = [&](size_t bytes) -> char* {
        off = (off + 255) & ~(size_t)255;
        char* p = ws + off;
        off += bytes;
        return p;
    };
    float*          GM   = (float*)alloc((size_t)8 * 256 * 4);
    float*          HH   = (float*)alloc((size_t)8 * 64 * 4);
    float*          WTS  = (float*)alloc((size_t)8 * 2304 * 4);
    unsigned short* WBa  = (unsigned short*)alloc((size_t)256 * 512 * 2);
    unsigned short* WBu1 = (unsigned short*)alloc((size_t)1024 * 9 * 256 * 2);
    unsigned short* WBu2 = (unsigned short*)alloc((size_t)128 * 256 * 2);
    unsigned short* WBr1 = (unsigned short*)alloc((size_t)32 * 9 * 128 * 2);
    unsigned short* WBr2 = (unsigned short*)alloc((size_t)128 * 9 * 32 * 2);
    char* R1 = alloc(67108864);   // T+ABIG -> CONV1 -> UPf
    char* R2 = alloc(16777216);   // FUSED  -> RE1
    char* R3 = alloc(67108864);   // SHUF
    char* R4 = alloc(33554432);   // UPB

    float*          T     = (float*)R1;                       // [8][256][4096] f32 NCHW
    unsigned short* ABIG  = (unsigned short*)(R1 + 33554432); // [8][4096][512] bf16 NHWC
    unsigned short* CONV1 = (unsigned short*)R1;              // [8][4096][1024] bf16 NHWC
    float*          UPf   = (float*)R1;                       // [8][16384][128] f32 NHWC
    unsigned short* FUSED = (unsigned short*)R2;              // [8][4096][256] bf16 NHWC
    unsigned short* RE1   = (unsigned short*)R2;              // [8][16384][32] bf16 NHWC
    unsigned short* SHUF  = (unsigned short*)R3;              // [8][16384][256] bf16 NHWC
    unsigned short* UPB   = (unsigned short*)R4;              // [8][16384][128] bf16 NHWC

    // ---- weight repack to bf16 [Cout][taps][Cin] ----
    auto wlaunch = [&](const float* s, unsigned short* d, int co, int ci, int tp) {
        const size_t n = (size_t)co * tp * ci;
        wprep_kernel<<<dim3((unsigned)((n + 255) / 256)), dim3(256), 0, stream>>>(s, d, co, ci, tp);
    };
    wlaunch(align_w, WBa,  256,  512, 1);
    wlaunch(up_w1,   WBu1, 1024, 256, 9);
    wlaunch(up_w2,   WBu2, 128,  256, 1);
    wlaunch(re_w1,   WBr1, 32,   128, 9);
    wlaunch(re_w2,   WBr2, 128,  32,  9);

    // ---- dyn_block(x1) and dyn_block(x2) -> concat (bf16 NHWC) in ABIG ----
    const float* xin[2] = { x1, x2 };
    for (int i = 0; i < 2; ++i) {
        // dsc1 + relu
        mean_hw_kernel<<<dim3(2048), dim3(256), 0, stream>>>(xin[i], GM);
        cond_fc1_gelu<<<dim3(2), dim3(256), 0, stream>>>(GM, dsc_w1[0], dsc_b1[0], HH);
        cond_fc2_softmax<<<dim3(8), dim3(256), 0, stream>>>(HH, dsc_w2[0], dsc_b2[0], WTS);
        dsc_apply<<<dim3(4096), dim3(256), 0, stream>>>(xin[i], WTS, T, nullptr, 0, 0, 1);
        // dsc2 -> bf16 NHWC slice of concat buffer
        mean_hw_kernel<<<dim3(2048), dim3(256), 0, stream>>>(T, GM);
        cond_fc1_gelu<<<dim3(2), dim3(256), 0, stream>>>(GM, dsc_w1[1], dsc_b1[1], HH);
        cond_fc2_softmax<<<dim3(8), dim3(256), 0, stream>>>(HH, dsc_w2[1], dsc_b2[1], WTS);
        dsc_apply<<<dim3(4096), dim3(256), 0, stream>>>(T, WTS, nullptr, ABIG, i * 256, 512, 0);
    }

    // ---- channel_align 1x1: 512 -> 256 @ 64x64 -> FUSED (bf16 NHWC) ----
    conv_wmma_nhwc<1, 4><<<dim3(64, 16, 8), dim3(32), 0, stream>>>(
        ABIG, WBa, align_b, nullptr, nullptr, nullptr, FUSED,
        512, 256, 4096, 6, 64, 0);

    // ---- upsampler conv 3x3: 256 -> 1024 @ 64x64 -> CONV1 (bf16 NHWC) ----
    conv_wmma_nhwc<9, 4><<<dim3(64, 64, 8), dim3(32), 0, stream>>>(
        FUSED, WBu1, up_b1, nullptr, nullptr, nullptr, CONV1,
        256, 1024, 4096, 6, 64, 0);

    // ---- pixel shuffle 2x (NHWC) ----
    pixshuf_nhwc<<<dim3(8192), dim3(256), 0, stream>>>(CONV1, SHUF);

    // ---- upsampler conv 1x1: 256 -> 128 @ 128x128 -> UPf (f32) + UPB (bf16) ----
    conv_wmma_nhwc<1, 4><<<dim3(256, 8, 8), dim3(32), 0, stream>>>(
        SHUF, WBu2, up_b2, nullptr, nullptr, UPf, UPB,
        256, 128, 16384, 7, 128, 0);

    // ---- res_enhance conv1 3x3 + ReLU: 128 -> 32 @ 128x128 -> RE1 ----
    conv_wmma_nhwc<9, 4><<<dim3(256, 2, 8), dim3(32), 0, stream>>>(
        UPB, WBr1, re_b1, nullptr, nullptr, nullptr, RE1,
        128, 32, 16384, 7, 128, 1);

    // ---- res_enhance conv2 3x3 + residual(up): 32 -> 128 -> d_out (NCHW) ----
    conv_wmma_nhwc<9, 4><<<dim3(256, 8, 8), dim3(32), 0, stream>>>(
        RE1, WBr2, re_b2, UPf, (float*)d_out, nullptr, nullptr,
        32, 128, 16384, 7, 128, 0);

    (void)in_sizes; (void)n_in; (void)out_size; (void)ws_size;
}
